// WindowAttention_54924041781628
// MI455X (gfx1250) — matrix operand order repaired and checked by reading, not compiled, not run on hardware
//
#include <hip/hip_runtime.h>

// Fused window-attention for MI455X (gfx1250, wave32, WMMA bf16 16x16x32).
// One workgroup (8 waves) processes WPB=8 windows sequentially; weights are
// staged/converted to bf16 once per block (8x amortization vs per-window).
// All intermediates live in LDS (~106KB -> 2 WGs/WGP with 320KB LDS).
// HBM traffic: x in (99.5MB) + y out (99.5MB) -> ~8.6us @ 23.3TB/s; bf16 WMMA
// keeps the ~18 GFLOP of GEMM work at/below that memory time.

#define WSZ   7
#define NPOS  49
#define CIN   62
#define CI    64    // inner channels (62 + 2 coords), padded K/M/N unit
#define CO3   192   // 3 * CI
#define STR   72    // LDS row stride (elements): 144B = 16B-aligned, bank-spread
#define HW    224
#define WPB   8     // windows per block
#define NBLK  1024  // 8192 windows / WPB

typedef __attribute__((ext_vector_type(16))) __bf16 v16bf;
typedef __attribute__((ext_vector_type(8)))  __bf16 v8bf;
typedef __attribute__((ext_vector_type(8)))  float  v8f;

union Pack2 { __bf16 b[2]; unsigned int u; };

// A fragment 16x32 bf16: lane(<16) row m0+r holds K = k0+[0..7] and k0+16+[0..7];
// lanes 16-31 hold K offset by +8. Source is row-major, K-contiguous, stride STR.
__device__ __forceinline__ v16bf load_a(const __bf16* base, int m0, int k0, int r, int half) {
    union { v16bf v; v8bf h[2]; } u;
    const __bf16* p = base + (m0 + r) * STR + k0 + 8 * half;
    u.h[0] = *(const v8bf*)(p);
    u.h[1] = *(const v8bf*)(p + 16);
    return u.v;
}
// B fragment 32x16 bf16: lanes 0-15 hold K=k0+[0..7] and k0+16+[0..7] of column n0+r,
// lanes 16-31 hold K offset by +8. Source is N-major, K-contiguous, stride STR.
__device__ __forceinline__ v16bf load_b(const __bf16* base, int n0, int k0, int r, int half) {
    union { v16bf v; v8bf h[2]; } u;
    const __bf16* p = base + (n0 + r) * STR + k0 + 8 * half;
    u.h[0] = *(const v8bf*)(p);
    u.h[1] = *(const v8bf*)(p + 16);
    return u.v;
}
// 16x16 output tile of a K=64 bf16 GEMM (two chained 16x16x32 WMMAs).
__device__ __forceinline__ v8f mm64(const __bf16* A, const __bf16* B, int m0, int n0,
                                    int r, int half) {
    v8f acc = {};
    acc = __builtin_amdgcn_wmma_f32_16x16x32_bf16(false, load_a(A, m0, 0, r, half),
                                                  false, load_b(B, n0, 0, r, half),
                                                  (short)0, acc, false, false);
    acc = __builtin_amdgcn_wmma_f32_16x16x32_bf16(false, load_a(A, m0, 32, r, half),
                                                  false, load_b(B, n0, 32, r, half),
                                                  (short)0, acc, false, false);
    return acc;
}

__global__ __launch_bounds__(256)
void win_attn_fused(const float* __restrict__ x, const float* __restrict__ w_qkv,
                    const float* __restrict__ w_proj, const float* __restrict__ b_proj,
                    float* __restrict__ out)
{
    __shared__ __bf16 sWq[CO3 * STR];  // A of GEMM1: w_qkv rows, K-contig
    __shared__ __bf16 sWp[CI  * STR];  // A of GEMM4: w_proj rows (rows 62-63 zero)
    __shared__ __bf16 sXt[CI  * STR];  // B of GEMM1: window input, pos-major (p x c)
    __shared__ __bf16 sQt[CI  * STR];  // A of GEMM2: Q^T * 0.125, pos-major (i x c)
    __shared__ __bf16 sKt[CI  * STR];  // B of GEMM2: K, pos-major (j x c)
    __shared__ __bf16 sVc[CI  * STR];  // A of GEMM3: V, ch-major (c x j)
    __shared__ __bf16 sP [CI  * STR];  // B of GEMM3: softmax probs (i x j)
    __shared__ __bf16 sOt[CI  * STR];  // B of GEMM4: attn out, pos-major (p x c)
    __shared__ float  sS [CI  * 64];   // attention logits (i x j), f32
    __shared__ float  sBp[CI];

    const int tid  = threadIdx.x;
    const int lane = tid & 31;
    const int wv   = __builtin_amdgcn_readfirstlane(tid >> 5);  // scalar wave id
    const int half = lane >> 4;
    const int r    = lane & 15;

    // ---- once per block: zero padded tiles, stage weights (fp32 -> bf16) ----
    for (int i = tid; i < CI * STR; i += 256) {
        sXt[i] = (__bf16)0.0f;
        sWp[i] = (__bf16)0.0f;
    }
    __syncthreads();
    for (int i = tid; i < CO3 * CI; i += 256) {
        int o = i >> 6, c = i & 63;
        sWq[o * STR + c] = (__bf16)w_qkv[i];
    }
    for (int i = tid; i < CIN * CI; i += 256) {
        int o = i >> 6, c = i & 63;
        sWp[o * STR + c] = (__bf16)w_proj[i];
    }
    if (tid < CI) sBp[tid] = (tid < CIN) ? b_proj[tid] : 0.0f;

    for (int win = 0; win < WPB; ++win) {
        const int wid = blockIdx.x * WPB + win;
        const int b   = wid >> 10;
        const int wh  = (wid >> 5) & 31;
        const int ww  = wid & 31;
        const int h0  = wh * WSZ;
        const int w0  = ww * WSZ;

        // ---- stage input window, pos-major, plus coord channels ----
        for (int i = tid; i < CIN * NPOS; i += 256) {
            int c = i / NPOS, p = i - c * NPOS;
            float v = x[(((size_t)b * CIN + c) * HW + (h0 + p / WSZ)) * HW + (w0 + p % WSZ)];
            sXt[p * STR + c] = (__bf16)v;
        }
        if (tid < NPOS) {
            int ph = tid / WSZ, pw = tid % WSZ;
            sXt[tid * STR + 62] = (__bf16)(-1.0f + 2.0f * (float)(w0 + pw) / 223.0f);
            sXt[tid * STR + 63] = (__bf16)(-1.0f + 2.0f * (float)(h0 + ph) / 223.0f);
        }
        // prefetch next window's x rows into cache while this window computes
        if (win + 1 < WPB) {
            const int nwid = wid + 1;
            const int nb = nwid >> 10, nwh = (nwid >> 5) & 31, nww = nwid & 31;
            for (int i = tid; i < CIN * WSZ; i += 256) {
                int c = i / WSZ, ph = i - c * WSZ;
                __builtin_prefetch(
                    &x[(((size_t)nb * CIN + c) * HW + (nwh * WSZ + ph)) * HW + nww * WSZ],
                    0, 3);
            }
        }
        __syncthreads();

        // ---- GEMM1: QKV[192x64] = Wq[192x64] x Xc[64x64(pos)]; 48 tiles, 6/wave ----
        for (int t = 0; t < 6; ++t) {
            int tile = wv * 6 + t;              // scalar
            int mt = tile >> 2, nt = tile & 3;  // scalar
            v8f acc = mm64(sWq, sXt, mt * 16, nt * 16, r, half);
            int n = nt * 16 + r;                // position
            if (mt < 4) {                       // Q -> pos-major, fold 1/sqrt(64)
                #pragma unroll
                for (int e = 0; e < 8; ++e) {
                    int m = mt * 16 + e + 8 * half;
                    sQt[n * STR + m] = (__bf16)(acc[e] * 0.125f);
                }
            } else if (mt < 8) {                // K -> pos-major
                #pragma unroll
                for (int e = 0; e < 8; ++e) {
                    int m = mt * 16 + e + 8 * half - 64;
                    sKt[n * STR + m] = (__bf16)acc[e];
                }
            } else {                            // V -> ch-major
                #pragma unroll
                for (int e = 0; e < 8; ++e) {
                    int m = mt * 16 + e + 8 * half - 128;
                    sVc[m * STR + n] = (__bf16)acc[e];
                }
            }
        }
        __syncthreads();

        // ---- GEMM2: S[i][j] = (Q^T K)*scale ; 16 tiles, 2/wave ----
        #pragma unroll
        for (int t = 0; t < 2; ++t) {
            int tile = wv + 8 * t;
            int mt = tile >> 2, nt = tile & 3;
            v8f acc = mm64(sQt, sKt, mt * 16, nt * 16, r, half);
            int j = nt * 16 + r;
            #pragma unroll
            for (int e = 0; e < 8; ++e) {
                int i = mt * 16 + e + 8 * half;
                sS[i * 64 + j] = acc[e];
            }
        }
        __syncthreads();

        // ---- softmax over j (mask padded j>=49); 8 rows per wave ----
        for (int rr = 0; rr < 8; ++rr) {
            int i  = wv * 8 + rr;               // scalar row
            int j0 = 2 * lane;
            float v0 = sS[i * 64 + j0];
            float v1 = sS[i * 64 + j0 + 1];
            if (j0     >= NPOS) v0 = -3.0e38f;
            if (j0 + 1 >= NPOS) v1 = -3.0e38f;
            float mx = fmaxf(v0, v1);
            #pragma unroll
            for (int off = 16; off >= 1; off >>= 1)
                mx = fmaxf(mx, __shfl_xor(mx, off, 32));
            float e0 = __expf(v0 - mx);
            float e1 = __expf(v1 - mx);
            float sm = e0 + e1;
            #pragma unroll
            for (int off = 16; off >= 1; off >>= 1)
                sm += __shfl_xor(sm, off, 32);
            float inv = 1.0f / sm;
            Pack2 pk;
            pk.b[0] = (__bf16)(e0 * inv);
            pk.b[1] = (__bf16)(e1 * inv);
            *(unsigned int*)&sP[i * STR + j0] = pk.u;  // packed b32, conflict-free
        }
        __syncthreads();

        // ---- GEMM3: Out[c][i] = sum_j V[c][j] P[i][j] ; 16 tiles, 2/wave ----
        #pragma unroll
        for (int t = 0; t < 2; ++t) {
            int tile = wv + 8 * t;
            int mt = tile >> 2, nt = tile & 3;
            v8f acc = mm64(sVc, sP, mt * 16, nt * 16, r, half);
            int n = nt * 16 + r;                // position i
            #pragma unroll
            for (int e = 0; e < 8; ++e) {
                int m = mt * 16 + e + 8 * half; // channel c
                sOt[n * STR + m] = (__bf16)acc[e];
            }
        }
        __syncthreads();

        // ---- GEMM4: Y = Wp[62x64] x Out[64x64(pos)] + bias, store to HBM ----
        #pragma unroll
        for (int t = 0; t < 2; ++t) {
            int tile = wv + 8 * t;
            int mt = tile >> 2, nt = tile & 3;
            v8f acc = mm64(sWp, sOt, mt * 16, nt * 16, r, half);
            int p = nt * 16 + r;
            if (p < NPOS) {
                int h = h0 + p / WSZ;
                int w = w0 + p % WSZ;
                #pragma unroll
                for (int e = 0; e < 8; ++e) {
                    int o = mt * 16 + e + 8 * half;
                    if (o < CIN)
                        out[(((size_t)b * CIN + o) * HW + h) * HW + w] = acc[e] + sBp[o];
                }
            }
        }
        __syncthreads();  // sOt/sP reuse safety before next window's GEMMs
    }
}

extern "C" void kernel_launch(void* const* d_in, const int* in_sizes, int n_in,
                              void* d_out, int out_size, void* d_ws, size_t ws_size,
                              hipStream_t stream) {
    (void)in_sizes; (void)n_in; (void)out_size; (void)d_ws; (void)ws_size;
    const float* x      = (const float*)d_in[0];
    const float* w_qkv  = (const float*)d_in[1];
    const float* w_proj = (const float*)d_in[2];
    const float* b_proj = (const float*)d_in[3];
    float* out = (float*)d_out;
    win_attn_fused<<<dim3(NBLK), dim3(256), 0, stream>>>(x, w_qkv, w_proj, b_proj, out);
}
